// MC_Att_L_19791209300070
// MI455X (gfx1250) — matrix operand order repaired and checked by reading, not compiled, not run on hardware
//
#include <hip/hip_runtime.h>
#include <hip/hip_bf16.h>

#define NN 100000
#define EE 1000000
#define FF 128
#define HH 128

typedef __attribute__((ext_vector_type(16))) __bf16 v16bf;
typedef __attribute__((ext_vector_type(8)))  float  v8f;
typedef __attribute__((ext_vector_type(4)))  unsigned int u32x4;

union Frag16 { v16bf v; __bf16 h[16]; u32x4 q[2]; };

__device__ __forceinline__ float warpSum(float s) {
#pragma unroll
  for (int off = 16; off > 0; off >>= 1) s += __shfl_xor(s, off, 32);
  return s;
}

// sign-aware float atomic max (works with -inf init)
__device__ __forceinline__ void atomicMaxF(float* addr, float val) {
  if (val >= 0.0f) atomicMax((int*)addr, __float_as_int(val));
  else atomicMin((unsigned int*)addr, __float_as_uint(val));
}

// ---------------- init kernels ----------------

__global__ void k_init_nodes(const float* __restrict__ h, float* __restrict__ hout,
                             __bf16* __restrict__ hb) {
  int i = blockIdx.x * 256 + threadIdx.x;
  if (i < NN * FF) { float x = h[i]; hout[i] = x; hb[i] = (__bf16)x; }
}

__global__ void k_init_misc(float* __restrict__ cagg, float* __restrict__ smax,
                            float* __restrict__ denom) {
  int i = blockIdx.x * 256 + threadIdx.x;
  if (i < NN * 9) cagg[i] = 0.0f;
  if (i < NN) { smax[i] = -3.4e38f; denom[i] = 0.0f; }
}

// Pack weights (bf16) into WMMA B-fragment order:
//   packed[((nt*4 + kk)*32 + lane)*16 + j] = W[(kk*32 + (lane<16?0:8) + (j&7) + (j>>3)*16)*ldn + nt*16 + (lane&15)]
// so each lane's 16 halfs are contiguous (two 16B vector loads per fragment).
__global__ void k_pack_w(const float* __restrict__ Wq, const float* __restrict__ Wkv,
                         const float* __restrict__ Wc1, __bf16* __restrict__ wqp,
                         __bf16* __restrict__ wkvp, __bf16* __restrict__ wc1p) {
  int idx = blockIdx.x * 256 + threadIdx.x;
  if (idx < 8 * 4 * 32 * 16) {       // 128-col matrices: Wq, Wc1
    int j = idx & 15, l = (idx >> 4) & 31, kk = (idx >> 9) & 3, nt = idx >> 11;
    int k = kk * 32 + ((l < 16) ? 0 : 8) + (j & 7) + ((j >> 3) << 4);
    int n = nt * 16 + (l & 15);
    wqp[idx]  = (__bf16)Wq[k * 128 + n];
    wc1p[idx] = (__bf16)Wc1[k * 128 + n];
  }
  if (idx < 16 * 4 * 32 * 16) {      // 256-col matrix: Wkv rows 9..136
    int j = idx & 15, l = (idx >> 4) & 31, kk = (idx >> 9) & 3, nt = idx >> 11;
    int k = kk * 32 + ((l < 16) ? 0 : 8) + (j & 7) + ((j >> 3) << 4);
    int n = nt * 16 + (l & 15);
    wkvp[idx] = (__bf16)Wkv[(9 + k) * 256 + n];
  }
}

// ---------------- node GEMMs (WMMA bf16) ----------------
// qn[N,128] = h @ Wq + bq
__global__ void k_gemm_qn(const __bf16* __restrict__ hb, const __bf16* __restrict__ wqp,
                          const float* __restrict__ bq, float* __restrict__ qn) {
  int mt = blockIdx.x;                 // 6250 M tiles
  int w  = threadIdx.x >> 5;           // 8 waves -> 8 N tiles
  int l  = threadIdx.x & 31;
  int half = l >> 4;
  int m16  = l & 15;
  int n = (w << 4) + m16;
  v8f acc = {};
#pragma unroll
  for (int kk = 0; kk < 4; ++kk) {
    Frag16 a, b;
    const __bf16* ap = hb + ((mt << 4) + m16) * FF + kk * 32 + (half << 3);
    a.q[0] = *(const u32x4*)(ap);        // K = base+0..7
    a.q[1] = *(const u32x4*)(ap + 16);   // K = base+16..23
    const __bf16* bp = wqp + ((((w << 2) + kk) << 5) + l) * 16;
    b.q[0] = *(const u32x4*)(bp);
    b.q[1] = *(const u32x4*)(bp + 8);
    acc = __builtin_amdgcn_wmma_f32_16x16x32_bf16(false, a.v, false, b.v,
                                                  (short)0, acc, false, false);
  }
  float bias = bq[n];
#pragma unroll
  for (int r = 0; r < 8; ++r) {
    int m = (mt << 4) + r + (half << 3);
    qn[m * HH + n] = acc[r] + bias;
  }
}

// kn/vn[N,128] = even/odd columns of h @ Wkv[9:,:] + bkv
__global__ void k_gemm_kvn(const __bf16* __restrict__ hb, const __bf16* __restrict__ wkvp,
                           const float* __restrict__ bkv, float* __restrict__ kn,
                           float* __restrict__ vn) {
  int mt = blockIdx.x;
  int w  = threadIdx.x >> 5;           // 16 waves -> 16 N tiles (256 cols)
  int l  = threadIdx.x & 31;
  int half = l >> 4;
  int m16  = l & 15;
  int n = (w << 4) + m16;
  v8f acc = {};
#pragma unroll
  for (int kk = 0; kk < 4; ++kk) {
    Frag16 a, b;
    const __bf16* ap = hb + ((mt << 4) + m16) * FF + kk * 32 + (half << 3);
    a.q[0] = *(const u32x4*)(ap);
    a.q[1] = *(const u32x4*)(ap + 16);
    const __bf16* bp = wkvp + ((((w << 2) + kk) << 5) + l) * 16;
    b.q[0] = *(const u32x4*)(bp);
    b.q[1] = *(const u32x4*)(bp + 8);
    acc = __builtin_amdgcn_wmma_f32_16x16x32_bf16(false, a.v, false, b.v,
                                                  (short)0, acc, false, false);
  }
  float bias = bkv[n];
#pragma unroll
  for (int r = 0; r < 8; ++r) {
    int m = (mt << 4) + r + (half << 3);
    float val = acc[r] + bias;
    if (n & 1) vn[m * HH + (n >> 1)] = val;
    else       kn[m * HH + (n >> 1)] = val;
  }
}

// qw[n,t] = qn[n] . Wkv[t, 0::2]   (t < 9)
__global__ void k_qw(const float* __restrict__ qn, const float* __restrict__ Wkv,
                     float* __restrict__ qw) {
  int i = blockIdx.x * 256 + threadIdx.x;
  if (i >= NN * 9) return;
  int node = i / 9, t = i - node * 9;
  const float* qp = qn + node * HH;
  const float* wp = Wkv + t * 256;
  float s = 0.0f;
#pragma unroll 8
  for (int k = 0; k < HH; ++k) s += qp[k] * wp[2 * k];
  qw[i] = s;
}

// ---------------- edge pass 1: alpha + segment max ----------------
__global__ void k_alpha(const int* __restrict__ row, const int* __restrict__ col,
                        const float* __restrict__ qn, const float* __restrict__ kn,
                        const float* __restrict__ qw, const float* __restrict__ coord,
                        float* __restrict__ alpha, float* __restrict__ smax) {
  int e = blockIdx.x * 8 + (threadIdx.x >> 5);
  if (e >= EE) return;
  int l = threadIdx.x & 31;
  int r = row[e], c = col[e];
  const float4* qp = (const float4*)(qn + r * HH);
  const float4* kp = (const float4*)(kn + c * HH);
  float4 qv = qp[l], kv = kp[l];
  float s = qv.x * kv.x + qv.y * kv.y + qv.z * kv.z + qv.w * kv.w;
  s = warpSum(s);
  if (l == 0) {
    float cd[9];
#pragma unroll
    for (int i = 0; i < 9; ++i) cd[i] = coord[r * 9 + i] - coord[c * 9 + i];
    float a = s;
#pragma unroll
    for (int i = 0; i < 3; ++i)
#pragma unroll
      for (int j = 0; j < 3; ++j) {
        float rad = cd[i*3]*cd[j*3] + cd[i*3+1]*cd[j*3+1] + cd[i*3+2]*cd[j*3+2];
        a += qw[r * 9 + i * 3 + j] * rad;
      }
    alpha[e] = a;
    atomicMaxF(&smax[r], a);
  }
}

// ---------------- edge pass 2: exp + denom ----------------
__global__ void k_exp(const int* __restrict__ row, float* __restrict__ alpha,
                      const float* __restrict__ smax, float* __restrict__ denom) {
  int e = blockIdx.x * 256 + threadIdx.x;
  if (e >= EE) return;
  int r = row[e];
  float ex = __expf(alpha[e] - smax[r]);
  alpha[e] = ex;                         // in-place: alpha now holds ex
  atomicAdd(&denom[r], ex);
}

// ---------------- edge pass 3: att, v, aggregation, coord path ----------------
__global__ void __launch_bounds__(256)
k_edge(const int* __restrict__ row, const int* __restrict__ col,
       const float* __restrict__ coord, const float* __restrict__ Wkv,
       const float* __restrict__ Wc2, const __bf16* __restrict__ wc1p,
       const float* __restrict__ vn, const float* __restrict__ ex,
       const float* __restrict__ denom, float* __restrict__ hout,
       float* __restrict__ cagg, float* __restrict__ attOut) {
  __shared__ int    s_row[16], s_col[16];
  __shared__ float  s_att[16], s_cd[16][9], s_rad[16][9], s_cv[16][3];
  __shared__ float  s_v[16][128];
  __shared__ __bf16 s_vb[16][128];
  __shared__ float  s_u[16][128];
  int tid = threadIdx.x;

  // (a) per-edge scalars
  if (tid < 16) {
    int e = blockIdx.x * 16 + tid;
    int r = row[e], c = col[e];
    s_row[tid] = r; s_col[tid] = c;
    float att = ex[e] / denom[r];
    s_att[tid] = att;
    attOut[e] = att;
    float cd[9];
#pragma unroll
    for (int i = 0; i < 9; ++i) { cd[i] = coord[r*9+i] - coord[c*9+i]; s_cd[tid][i] = cd[i]; }
#pragma unroll
    for (int i = 0; i < 3; ++i)
#pragma unroll
      for (int j = 0; j < 3; ++j)
        s_rad[tid][i*3+j] = cd[i*3]*cd[j*3] + cd[i*3+1]*cd[j*3+1] + cd[i*3+2]*cd[j*3+2];
  }
  __syncthreads();

  // (b) v = vn[col] + radial @ Wkv[:9, 1::2]; h aggregation atomics; bf16 copy for WMMA
  for (int idx = tid; idx < 16 * 128; idx += 256) {
    int e16 = idx >> 7, n = idx & 127;
    int c = s_col[e16];
    float val = vn[c * HH + n];
#pragma unroll
    for (int t = 0; t < 9; ++t) val += s_rad[e16][t] * Wkv[t * 256 + 2 * n + 1];
    s_v[e16][n] = val;
    s_vb[e16][n] = (__bf16)val;
    atomicAdd(&hout[s_row[e16] * HH + n], s_att[e16] * val);
  }
  __syncthreads();

  // (c) u = silu(v @ Wc1)  — 16x128x128 WMMA, A from LDS bf16, B pre-packed
  {
    int w = tid >> 5, l = tid & 31;
    int half = l >> 4, m16 = l & 15;
    int n = (w << 4) + m16;
    v8f acc = {};
#pragma unroll
    for (int kk = 0; kk < 4; ++kk) {
      Frag16 a, b;
      const __bf16* ap = &s_vb[m16][kk * 32 + (half << 3)];
      a.q[0] = *(const u32x4*)(ap);
      a.q[1] = *(const u32x4*)(ap + 16);
      const __bf16* bp = wc1p + ((((w << 2) + kk) << 5) + l) * 16;
      b.q[0] = *(const u32x4*)(bp);
      b.q[1] = *(const u32x4*)(bp + 8);
      acc = __builtin_amdgcn_wmma_f32_16x16x32_bf16(false, a.v, false, b.v,
                                                    (short)0, acc, false, false);
    }
#pragma unroll
    for (int r = 0; r < 8; ++r) {
      float x = acc[r];
      s_u[r + (half << 3)][n] = x / (1.0f + __expf(-x));
    }
  }
  __syncthreads();

  // (d) coord_v = att * (u @ Wc2)
  if (tid < 48) {
    int e16 = tid / 3, ci = tid - e16 * 3;
    float s = 0.0f;
#pragma unroll 8
    for (int n = 0; n < 128; ++n) s += s_u[e16][n] * Wc2[n * 3 + ci];
    s_cv[e16][ci] = s_att[e16] * s;
  }
  __syncthreads();

  // (e) trans aggregation
  if (tid < 144) {
    int e16 = tid / 9, t = tid - e16 * 9, i = t / 3;
    atomicAdd(&cagg[s_row[e16] * 9 + t], s_cd[e16][t] * s_cv[e16][i]);
  }
}

// ---------------- final coord update ----------------
__global__ void k_coord(const float* __restrict__ coord, const float* __restrict__ cagg,
                        float* __restrict__ coordOut) {
  int i = blockIdx.x * 256 + threadIdx.x;
  if (i >= NN * 9) return;
  float v = cagg[i];
  v = fminf(10.0f, fmaxf(-10.0f, v));
  coordOut[i] = coord[i] + v;
}

extern "C" void kernel_launch(void* const* d_in, const int* in_sizes, int n_in,
                              void* d_out, int out_size, void* d_ws, size_t ws_size,
                              hipStream_t stream) {
  const float* h     = (const float*)d_in[0];
  const float* coord = (const float*)d_in[1];
  const int*   row   = (const int*)d_in[2];
  const int*   col   = (const int*)d_in[3];
  const float* Wq    = (const float*)d_in[4];
  const float* bq    = (const float*)d_in[5];
  const float* Wkv   = (const float*)d_in[6];
  const float* bkv   = (const float*)d_in[7];
  const float* Wc1   = (const float*)d_in[8];
  const float* Wc2   = (const float*)d_in[9];

  float* out      = (float*)d_out;
  float* hout     = out;                       // [N,128]
  float* coordOut = out + NN * 128;            // [N,9]
  float* attOut   = out + NN * 128 + NN * 9;   // [E]

  char* ws = (char*)d_ws;
  size_t off = 0;
  auto alloc = [&](size_t bytes) { char* p = ws + off; off += (bytes + 15) & ~size_t(15); return p; };
  float*  qn     = (float*)alloc(sizeof(float) * NN * 128);
  float*  kn     = (float*)alloc(sizeof(float) * NN * 128);
  float*  vn     = (float*)alloc(sizeof(float) * NN * 128);
  float*  qw     = (float*)alloc(sizeof(float) * NN * 9);
  float*  alpha  = (float*)alloc(sizeof(float) * EE);
  float*  smax   = (float*)alloc(sizeof(float) * NN);
  float*  denom  = (float*)alloc(sizeof(float) * NN);
  float*  cagg   = (float*)alloc(sizeof(float) * NN * 9);
  __bf16* hb     = (__bf16*)alloc(sizeof(__bf16) * NN * 128);
  __bf16* wqp    = (__bf16*)alloc(sizeof(__bf16) * 128 * 128);
  __bf16* wkvp   = (__bf16*)alloc(sizeof(__bf16) * 128 * 256);
  __bf16* wc1p   = (__bf16*)alloc(sizeof(__bf16) * 128 * 128);

  k_init_nodes<<<(NN * 128 + 255) / 256, 256, 0, stream>>>(h, hout, hb);
  k_init_misc <<<(NN * 9 + 255) / 256, 256, 0, stream>>>(cagg, smax, denom);
  k_pack_w    <<<(128 * 256 + 255) / 256, 256, 0, stream>>>(Wq, Wkv, Wc1, wqp, wkvp, wc1p);

  k_gemm_qn <<<NN / 16, 256, 0, stream>>>(hb, wqp, bq, qn);
  k_gemm_kvn<<<NN / 16, 512, 0, stream>>>(hb, wkvp, bkv, kn, vn);
  k_qw      <<<(NN * 9 + 255) / 256, 256, 0, stream>>>(qn, Wkv, qw);

  k_alpha<<<EE / 8, 256, 0, stream>>>(row, col, qn, kn, qw, coord, alpha, smax);
  k_exp  <<<(EE + 255) / 256, 256, 0, stream>>>(row, alpha, smax, denom);
  k_edge <<<EE / 16, 256, 0, stream>>>(row, col, coord, Wkv, Wc2, wc1p, vn,
                                       alpha, denom, hout, cagg, attOut);
  k_coord<<<(NN * 9 + 255) / 256, 256, 0, stream>>>(coord, cagg, coordOut);
}
